// GRU_90391881712032
// MI455X (gfx1250) — matrix operand order repaired and checked by reading, not compiled, not run on hardware
//
#include <hip/hip_runtime.h>

// Problem constants (match reference)
#define VV 32000
#define EE 512
#define HH 1024
#define GG (3 * HH)
#define BB 16
#define TT 256

typedef __attribute__((ext_vector_type(16))) __bf16 v16bf;
typedef __attribute__((ext_vector_type(8)))  __bf16 bf16x8;
typedef __attribute__((ext_vector_type(8)))  float  v8f;

static __device__ __forceinline__ __bf16 f2b(float f) { return (__bf16)f; }

// Build a 16-bit A/B-fragment from two contiguous 8-element chunks
static __device__ __forceinline__ v16bf frag_from2(const __bf16* p) {
  bf16x8 lo = *(const bf16x8*)(p);
  bf16x8 hi = *(const bf16x8*)(p + 16);
  v16bf f;
#pragma unroll
  for (int i = 0; i < 8; ++i) { f[i] = lo[i]; f[i + 8] = hi[i]; }
  return f;
}

// ---------------- prep kernels ----------------

// out[C,R] (bf16, K-major for WMMA B) = transpose(in[R,C] fp32)
__global__ void k_transpose_bf16(const float* __restrict__ in, __bf16* __restrict__ out,
                                 int R, int C) {
  int i = blockIdx.x * blockDim.x + threadIdx.x;
  if (i < R * C) {
    int r = i / C, c = i - r * C;
    out[(size_t)c * R + r] = f2b(in[i]);
  }
}

// xe[T,B,E] bf16 = emb[x[b,t]]   (x is [B,T], int32 token ids)
__global__ void k_embed(const int* __restrict__ x, const float* __restrict__ emb,
                        __bf16* __restrict__ xe) {
  int i = blockIdx.x * blockDim.x + threadIdx.x;
  if (i >= TT * BB * EE) return;
  int e  = i % EE;
  int tb = i / EE;
  int b  = tb % BB;
  int t  = tb / BB;
  int tok = x[b * TT + t];
  xe[i] = f2b(emb[(size_t)tok * EE + e]);
}

__global__ void k_init_h(const float* __restrict__ src, float* __restrict__ dF,
                         __bf16* __restrict__ dB, int n) {
  int i = blockIdx.x * blockDim.x + threadIdx.x;
  if (i < n) { float v = src[i]; dF[i] = v; dB[i] = f2b(v); }
}

// ---------------- bf16 WMMA GEMM + bias ----------------
// C[M,N] fp32 = A[M,K] bf16 (row-major) x W[K,N] bf16 (K-major) + bias[N]
// grid = (N/256, M/32), block = 128 (4 waves).
// Each wave: 32 rows x 64 cols = 2x4 WMMA tiles -> 8 independent v_wmma per K-step,
// B fragments reused across both row tiles (halves B traffic, hides load latency).
__global__ __launch_bounds__(128)
void k_gemm_bias(const __bf16* __restrict__ A, const __bf16* __restrict__ W,
                 const float* __restrict__ bias, float* __restrict__ C,
                 int N, int K) {
  const int lane = threadIdx.x & 31;
  const int wave = threadIdx.x >> 5;
  const int hi   = lane >> 4;   // lane half selects K sub-group (per CDNA5 A layout)
  const int nn   = lane & 15;
  const int m0   = blockIdx.y << 5;             // 32 rows per block
  const int nb   = blockIdx.x * 256 + wave * 64;

  // loop-carried pointers (pure pointer-bump addressing in the inner loop)
  const __bf16* ap0 = A + (size_t)(m0 + nn) * K + 8 * hi;        // row tile 0
  const __bf16* ap1 = ap0 + (size_t)16 * K;                      // row tile 1
  const __bf16* bp  = W + (size_t)(8 * hi) * N + nb + nn;        // B col N=nn, K base 8*hi
  const size_t  bstep = (size_t)32 * N;

  v8f acc[2][4] = {};
  for (int k = 0; k < K; k += 32) {
    if (k + 32 < K)  // prefetch next K-slab of W (global_prefetch_b8, speculative)
      __builtin_prefetch((const void*)(bp + bstep), 0, 1);

    // issue ALL loads for this K-step first, then 8 independent WMMAs
    v16bf a0 = frag_from2(ap0);
    v16bf a1 = frag_from2(ap1);
    v16bf b[4];
#pragma unroll
    for (int s = 0; s < 4; ++s) {
#pragma unroll
      for (int e = 0; e < 8; ++e) { b[s][e] = bp[(size_t)e * N + 16 * s]; b[s][e + 8] = bp[(size_t)(16 + e) * N + 16 * s]; }
    }

#pragma unroll
    for (int s = 0; s < 4; ++s) {
      acc[0][s] = __builtin_amdgcn_wmma_f32_16x16x32_bf16(false, a0, false, b[s],
                                                          (short)0, acc[0][s], false, false);
      acc[1][s] = __builtin_amdgcn_wmma_f32_16x16x32_bf16(false, a1, false, b[s],
                                                          (short)0, acc[1][s], false, false);
    }
    ap0 += 32; ap1 += 32; bp += bstep;
  }

  // D layout: VGPR i, lane -> (M = i + 8*hi, N = lane&15)
#pragma unroll
  for (int s = 0; s < 4; ++s) {
    int n = nb + 16 * s + nn;
    float bv = bias ? bias[n] : 0.f;
#pragma unroll
    for (int r = 0; r < 2; ++r) {
#pragma unroll
      for (int i = 0; i < 8; ++i) {
        C[(size_t)(m0 + 16 * r + i + 8 * hi) * N + n] = acc[r][s][i] + bv;
      }
    }
  }
}

// ---------------- GRU recurrent step ----------------
// One launch per timestep. grid = H/16 blocks (16 H-columns each), block = 256 (8 waves).
// Each wave accumulates K-chunk of 128 for the 3 gate column-groups; LDS reduce; gates.
__global__ __launch_bounds__(256)
void k_gru_step(const float* __restrict__ xg,      // [T,B,3H] fp32 (b_ih already added)
                const float* __restrict__ b_hh,    // [3H]
                const __bf16* __restrict__ wt,     // w_hh^T, K-major [H, 3H] bf16
                const float*  __restrict__ hF_in,  // [B,H] fp32
                const __bf16* __restrict__ hB_in,  // [B,H] bf16
                float*  __restrict__ hF_out,
                __bf16* __restrict__ hB_out,
                __bf16* __restrict__ y,            // output seq (layout via strides)
                int t, int ys_t, int ys_b) {
  __shared__ float red[3 * 8 * 16 * 16];           // 24 KB
  const int lane = threadIdx.x & 31;
  const int wave = threadIdx.x >> 5;
  const int hi   = lane >> 4;
  const int nn   = lane & 15;
  const int j0   = blockIdx.x << 4;

  const int kbase = wave * 128;
  const __bf16* ap = hB_in + nn * HH + kbase + 8 * hi;           // row = batch index
  const __bf16* bp = wt + (size_t)(kbase + 8 * hi) * GG + j0 + nn;
  const size_t  bstep = (size_t)32 * GG;

  v8f ar = {}, az = {}, an = {};
#pragma unroll
  for (int s = 0; s < 4; ++s) {
    v16bf a  = frag_from2(ap);
    v16bf br, bz, bn;
#pragma unroll
    for (int e = 0; e < 8; ++e) {
      br[e] = bp[(size_t)e * GG]; br[e + 8] = bp[(size_t)(16 + e) * GG];
      bz[e] = bp[(size_t)e * GG + HH]; bz[e + 8] = bp[(size_t)(16 + e) * GG + HH];
      bn[e] = bp[(size_t)e * GG + 2 * HH]; bn[e + 8] = bp[(size_t)(16 + e) * GG + 2 * HH];
    }
    ar = __builtin_amdgcn_wmma_f32_16x16x32_bf16(false, a, false, br, (short)0, ar, false, false);
    az = __builtin_amdgcn_wmma_f32_16x16x32_bf16(false, a, false, bz, (short)0, az, false, false);
    an = __builtin_amdgcn_wmma_f32_16x16x32_bf16(false, a, false, bn, (short)0, an, false, false);
    ap += 32; bp += bstep;
  }
#pragma unroll
  for (int i = 0; i < 8; ++i) {
    int m = i + 8 * hi;
    red[((0 * 8 + wave) * 16 + m) * 16 + nn] = ar[i];
    red[((1 * 8 + wave) * 16 + m) * 16 + nn] = az[i];
    red[((2 * 8 + wave) * 16 + m) * 16 + nn] = an[i];
  }
  __syncthreads();

  // 256 threads = one per (batch b, column jj); reduce the 8 K-partials, apply gates
  const int b  = threadIdx.x >> 4;
  const int jj = threadIdx.x & 15;
  float sr = 0.f, sz = 0.f, sn = 0.f;
#pragma unroll
  for (int w = 0; w < 8; ++w) {
    sr += red[((0 * 8 + w) * 16 + b) * 16 + jj];
    sz += red[((1 * 8 + w) * 16 + b) * 16 + jj];
    sn += red[((2 * 8 + w) * 16 + b) * 16 + jj];
  }
  int j = j0 + jj;
  const float* xgr = xg + ((size_t)t * BB + b) * GG;
  float xr = xgr[j], xz = xgr[HH + j], xn = xgr[2 * HH + j];
  float hr = sr + b_hh[j];
  float hz = sz + b_hh[HH + j];
  float hn = sn + b_hh[2 * HH + j];
  float r  = 1.f / (1.f + __expf(-(xr + hr)));
  float z  = 1.f / (1.f + __expf(-(xz + hz)));
  float ng = tanhf(xn + r * hn);
  float hold = hF_in[b * HH + j];
  float hnew = (1.f - z) * ng + z * hold;
  hF_out[b * HH + j] = hnew;
  hB_out[b * HH + j] = f2b(hnew);
  y[(size_t)t * ys_t + (size_t)b * ys_b + j] = f2b(hnew);
}

// ---------------- host driver ----------------
extern "C" void kernel_launch(void* const* d_in, const int* in_sizes, int n_in,
                              void* d_out, int out_size, void* d_ws, size_t ws_size,
                              hipStream_t stream) {
  (void)in_sizes; (void)n_in; (void)out_size; (void)ws_size;

  const int*   x      = (const int*)  d_in[0];
  const float* hidden = (const float*)d_in[1];
  const float* emb    = (const float*)d_in[2];
  const float* w_ih0  = (const float*)d_in[3];
  const float* w_hh0  = (const float*)d_in[4];
  const float* b_ih0  = (const float*)d_in[5];
  const float* b_hh0  = (const float*)d_in[6];
  const float* w_ih1  = (const float*)d_in[7];
  const float* w_hh1  = (const float*)d_in[8];
  const float* b_ih1  = (const float*)d_in[9];
  const float* b_hh1  = (const float*)d_in[10];
  const float* fc_w   = (const float*)d_in[11];
  const float* fc_b   = (const float*)d_in[12];
  float* out = (float*)d_out;

  // workspace carve (~160 MB), 256B aligned
  char* base = (char*)d_ws; size_t off = 0;
  auto alloc = [&](size_t bytes) -> void* {
    void* r = base + off;
    off = (off + bytes + 255) & ~(size_t)255;
    return r;
  };
  __bf16* wt_ih0 = (__bf16*)alloc((size_t)EE * GG * 2);
  __bf16* wt_hh0 = (__bf16*)alloc((size_t)HH * GG * 2);
  __bf16* wt_ih1 = (__bf16*)alloc((size_t)HH * GG * 2);
  __bf16* wt_hh1 = (__bf16*)alloc((size_t)HH * GG * 2);
  __bf16* wt_fc  = (__bf16*)alloc((size_t)HH * VV * 2);
  __bf16* xe     = (__bf16*)alloc((size_t)TT * BB * EE * 2);
  float*  xg     = (float*) alloc((size_t)TT * BB * GG * 4);   // reused by both layers
  __bf16* y0     = (__bf16*)alloc((size_t)TT * BB * HH * 2);   // [T,B,H]
  __bf16* y1     = (__bf16*)alloc((size_t)TT * BB * HH * 2);   // [B,T,H]
  float*  h0F[2] = { (float*)alloc(BB * HH * 4), (float*)alloc(BB * HH * 4) };
  float*  h1F[2] = { (float*)alloc(BB * HH * 4), (float*)alloc(BB * HH * 4) };
  __bf16* h0B[2] = { (__bf16*)alloc(BB * HH * 2), (__bf16*)alloc(BB * HH * 2) };
  __bf16* h1B[2] = { (__bf16*)alloc(BB * HH * 2), (__bf16*)alloc(BB * HH * 2) };

  // 1) weights -> bf16, K-major
  auto tp = [&](const float* w, __bf16* wt, int R, int C) {
    int tot = R * C;
    k_transpose_bf16<<<(tot + 255) / 256, 256, 0, stream>>>(w, wt, R, C);
  };
  tp(w_ih0, wt_ih0, GG, EE);
  tp(w_hh0, wt_hh0, GG, HH);
  tp(w_ih1, wt_ih1, GG, HH);
  tp(w_hh1, wt_hh1, GG, HH);
  tp(fc_w,  wt_fc,  VV, HH);

  // 2) embedding gather -> bf16 [T,B,E]
  k_embed<<<(TT * BB * EE + 255) / 256, 256, 0, stream>>>(x, emb, xe);

  // 3) init hidden (double-buffered)
  k_init_h<<<(BB * HH + 255) / 256, 256, 0, stream>>>(hidden,            h0F[0], h0B[0], BB * HH);
  k_init_h<<<(BB * HH + 255) / 256, 256, 0, stream>>>(hidden + BB * HH,  h1F[0], h1B[0], BB * HH);

  dim3 gxg(GG / 256, (TT * BB) / 32);

  // 4) layer 0: xg = xe @ w_ih0^T + b_ih0, then 256 recurrent steps
  k_gemm_bias<<<gxg, 128, 0, stream>>>(xe, wt_ih0, b_ih0, xg, GG, EE);
  for (int t = 0; t < TT; ++t) {
    int pi = t & 1, po = pi ^ 1;
    k_gru_step<<<HH / 16, 256, 0, stream>>>(xg, b_hh0, wt_hh0,
        h0F[pi], h0B[pi], h0F[po], h0B[po],
        y0, t, BB * HH, HH);                       // y0 time-major [T,B,H]
  }

  // 5) layer 1: xg = y0 @ w_ih1^T + b_ih1, then 256 recurrent steps
  k_gemm_bias<<<gxg, 128, 0, stream>>>(y0, wt_ih1, b_ih1, xg, GG, HH);
  for (int t = 0; t < TT; ++t) {
    int pi = t & 1, po = pi ^ 1;
    k_gru_step<<<HH / 16, 256, 0, stream>>>(xg, b_hh1, wt_hh1,
        h1F[pi], h1B[pi], h1F[po], h1B[po],
        y1, t, HH, TT * HH);                       // y1 batch-major [B,T,H]
  }

  // 6) prediction = y1 @ fc_w^T + fc_b  -> d_out [B,T,V]
  dim3 gfc(VV / 256, (TT * BB) / 32);
  k_gemm_bias<<<gfc, 128, 0, stream>>>(y1, wt_fc, fc_b, out, VV, HH);

  // 7) hidden_out = final h (T even -> final state lives in buffer index 0)
  size_t predN = (size_t)BB * TT * VV;
  int fin = TT & 1;
  hipMemcpyAsync(out + predN,           h0F[fin], BB * HH * sizeof(float),
                 hipMemcpyDeviceToDevice, stream);
  hipMemcpyAsync(out + predN + BB * HH, h1F[fin], BB * HH * sizeof(float),
                 hipMemcpyDeviceToDevice, stream);
}
